// ContrastMemory_55370718380635
// MI455X (gfx1250) — matrix operand order repaired and checked by reading, not compiled, not run on hardware
//
#include <hip/hip_runtime.h>
#include <float.h>
#include <math.h>

typedef __attribute__((ext_vector_type(16))) __bf16 v16bf;
typedef __attribute__((ext_vector_type(8)))  float  v8f;
typedef __attribute__((ext_vector_type(4)))  float  f4v;

#define B_SZ    128
#define D_DIM   128
#define N_MEM   100000
#define K_FULL  8194          // POS_K+1 + NEG_K
#define N_TILES 513           // ceil(8194/16)
#define SPLIT   8             // k-chunks (blocks) per batch
#define INV_TAU (1.0f/0.07f)

// ---------------------------------------------------------------- helpers ---

__device__ inline v8f wmma_bf16(v16bf a, v16bf b, v8f c) {
  // D = A(16x32 bf16) * B(32x16 bf16) + C(16x16 f32)
  return __builtin_amdgcn_wmma_f32_16x16x32_bf16(false, a, false, b,
                                                 (short)0, c, false, false);
}

__device__ inline void split_set(v16bf& hi, v16bf& lo, int i, float x) {
  __bf16 h = (__bf16)x;
  hi[i] = h;
  lo[i] = (__bf16)(x - (float)h);
}

// 16 f32 (two 8-float chunks pa[0..7], pb[0..7]) -> bf16 hi/lo vectors
__device__ inline void split16(const float* pa, const float* pb,
                               v16bf& hi, v16bf& lo) {
  f4v c0 = *(const f4v*)pa;
  f4v c1 = *(const f4v*)(pa + 4);
  f4v c2 = *(const f4v*)pb;
  f4v c3 = *(const f4v*)(pb + 4);
#pragma unroll
  for (int i = 0; i < 4; ++i) {
    split_set(hi, lo, i,      c0[i]);
    split_set(hi, lo, 4 + i,  c1[i]);
    split_set(hi, lo, 8 + i,  c2[i]);
    split_set(hi, lo, 12 + i, c3[i]);
  }
}

__device__ inline void zero16(v16bf& hi, v16bf& lo) {
#pragma unroll
  for (int i = 0; i < 16; ++i) { hi[i] = (__bf16)0.0f; lo[i] = (__bf16)0.0f; }
}

// online softmax state update for value v (teacher side cross-sum W against `other`)
__device__ inline void online_upd(float v, float other,
                                  float& M, float& Z1, float& Z3, float& W) {
  if (v > M) {
    float d  = M - v;              // <= 0
    Z1 *= __expf(d);
    float c3 = __expf(d * (1.0f/3.0f));
    Z3 *= c3; W *= c3;
    M = v;
  }
  float d  = v - M;
  Z1 += __expf(d);
  float e3 = __expf(d * (1.0f/3.0f));
  Z3 += e3;
  W  += e3 * (v - other);
}

// --------------------------------------------------- fused logits kernel ----
// grid = B_SZ*SPLIT blocks x 256 threads (8 waves). Block (b, chunk) handles a
// strided subset of the 513 k-tiles of batch b. Per 16-k tile, each wave:
//   A0 = 16 gathered mem0 rows, A1 = 16 gathered mem1 rows (hi/lo bf16 split)
//   B0 cols: {emb0[b], emb1[b], 0...},  B1 cols: {emb1[b], emb0[b], 0...}
//   D0 = A0*B0, D1 = A1*B1  via 3-term bf16 WMMA (hi*hi + hi*lo + lo*hi)
// Lanes 0/16 then own the intra pair (a0,a1), lanes 1/17 the inter pair
// (c01,c10), and run online max/sumexp/cross-sum reductions.
__global__ __launch_bounds__(256) void cm_logits_kernel(
    const float* __restrict__ emb0, const float* __restrict__ emb1,
    const float* __restrict__ mem0, const float* __restrict__ mem1,
    const int* __restrict__ pos_idx, const int* __restrict__ neg_idx,
    float* __restrict__ chunk_state,   // [B][SPLIT][16]
    float* __restrict__ heads_out)     // [B][4]
{
  const int b     = blockIdx.x >> 3;
  const int chunk = blockIdx.x & (SPLIT - 1);
  const int tid   = threadIdx.x;
  const int wave  = tid >> 5;
  const int lane  = tid & 31;
  const int half  = lane >> 4;    // K-half selector for A/B layouts
  const int ncol  = lane & 15;    // matrix row (A) / column (B,D) index

  const float* e0 = emb0 + b * D_DIM;
  const float* e1 = emb1 + b * D_DIM;
  const int* pi = pos_idx + b * 2;
  const int* ni = neg_idx + b * 8192;

  // B column sources: bank0 col0=emb0,col1=emb1; bank1 swapped
  const float* srcB0 = (ncol == 0) ? e0 : ((ncol == 1) ? e1 : (const float*)0);
  const float* srcB1 = (ncol == 0) ? e1 : ((ncol == 1) ? e0 : (const float*)0);

  // loop-invariant B tiles, converted once into LDS: [kstep][lane][4x16 bf16]
  __shared__ __align__(16) __bf16 Blds[4][32][64];
  __shared__ float st[2][16][8];

  if (wave == 0) {
#pragma unroll
    for (int s = 0; s < 4; ++s) {
      const int off = s * 32 + half * 16;   // contiguous 16 K-values per lane
      v16bf bh, bl;
      if (srcB0) split16(srcB0 + off, srcB0 + off + 8, bh, bl);
      else       zero16(bh, bl);
      *(v16bf*)&Blds[s][lane][0]  = bh;
      *(v16bf*)&Blds[s][lane][16] = bl;
      if (srcB1) split16(srcB1 + off, srcB1 + off + 8, bh, bl);
      else       zero16(bh, bl);
      *(v16bf*)&Blds[s][lane][32] = bh;
      *(v16bf*)&Blds[s][lane][48] = bl;
    }
  }
  __syncthreads();

  // online state (only meaningful for lanes with ncol < 2)
  float Mx = -FLT_MAX, Z1x = 0.f, Z3x = 0.f, Wx = 0.f;   // x = bank0 vector
  float My = -FLT_MAX, Z1y = 0.f, Z3y = 0.f, Wy = 0.f;   // y = bank1 vector
  float h0 = 0.f, h1 = 0.f;                              // head terms

  const int baseA = half * 16;
  const int gwave = chunk * 8 + wave;                    // 0..63

  for (int t = gwave; t < N_TILES; t += 64) {
    // gather: A-matrix row M = ncol of this tile
    int krow = t * 16 + ncol;
    int kc   = krow < (K_FULL - 1) ? krow : (K_FULL - 1);
    int ridx = (kc < 2) ? pi[kc] : ni[kc - 2];
    const float* r0 = mem0 + (size_t)ridx * D_DIM;
    const float* r1 = mem1 + (size_t)ridx * D_DIM;

    // prefetch the NEXT tile's gathered rows into near caches while this
    // tile's loads/WMMAs run (global_prefetch_b8; rows are L2-resident, this
    // hides L2->L0 latency on the random gather). Lane pair (ncol, ncol+16)
    // covers all four 128B lines of each 512B row.
    {
      int tn = t + 64;
      if (tn < N_TILES) {
        int krown = tn * 16 + ncol;
        int kcn   = krown < (K_FULL - 1) ? krown : (K_FULL - 1);
        int ridxn = (kcn < 2) ? pi[kcn] : ni[kcn - 2];
        const float* p0 = mem0 + (size_t)ridxn * D_DIM + half * 64;
        const float* p1 = mem1 + (size_t)ridxn * D_DIM + half * 64;
        __builtin_prefetch(p0, 0, 3);
        __builtin_prefetch(p0 + 32, 0, 3);
        __builtin_prefetch(p1, 0, 3);
        __builtin_prefetch(p1 + 32, 0, 3);
      }
    }

    v8f acc0 = {};
    v8f acc1 = {};

#pragma unroll
    for (int s = 0; s < 4; ++s) {
      const int d0 = s * 32;
      v16bf a0h, a0l, a1h, a1l;
      split16(r0 + d0 + baseA, r0 + d0 + baseA + 8, a0h, a0l);
      split16(r1 + d0 + baseA, r1 + d0 + baseA + 8, a1h, a1l);
      v16bf b0h = *(const v16bf*)&Blds[s][lane][0];
      v16bf b0l = *(const v16bf*)&Blds[s][lane][16];
      v16bf b1h = *(const v16bf*)&Blds[s][lane][32];
      v16bf b1l = *(const v16bf*)&Blds[s][lane][48];

      acc0 = wmma_bf16(a0h, b0h, acc0);
      acc0 = wmma_bf16(a0h, b0l, acc0);
      acc0 = wmma_bf16(a0l, b0h, acc0);
      acc1 = wmma_bf16(a1h, b1h, acc1);
      acc1 = wmma_bf16(a1h, b1l, acc1);
      acc1 = wmma_bf16(a1l, b1h, acc1);
    }

    // D layout: lane (ncol + 16*half) vgpr r -> element (m = r + 8*half, n = ncol)
    if (ncol < 2) {
#pragma unroll
      for (int r = 0; r < 8; ++r) {
        int k = t * 16 + half * 8 + r;
        if (k < K_FULL) {
          float x = acc0[r] * INV_TAU;
          float y = acc1[r] * INV_TAU;
          if (ncol == 0) {                    // intra pair: x=a0, y=a1 (k>=1)
            if (k != 0) {
              if (k == 1) { h0 = x; h1 = y; } // first intra logit
              online_upd(x, y, Mx, Z1x, Z3x, Wx);
              online_upd(y, x, My, Z1y, Z3y, Wy);
            }
          } else {                            // inter pair: x=c01, y=c10
            if (k < 2) { h0 += x; h1 += y; }  // npos=2 head sum
            online_upd(x, y, Mx, Z1x, Z3x, Wx);
            online_upd(y, x, My, Z1y, Z3y, Wy);
          }
        }
      }
    }
  }

  if (ncol < 2) {
    float* p = st[ncol][wave * 2 + half];
    p[0] = Mx; p[1] = Z1x; p[2] = Z3x; p[3] = Wx;
    p[4] = My; p[5] = Z1y; p[6] = Z3y; p[7] = Wy;
    if (gwave == 0 && half == 0) {
      heads_out[b * 4 + ncol * 2 + 0] = h0;
      heads_out[b * 4 + ncol * 2 + 1] = h1;
    }
  }
  __syncthreads();

  // deterministic in-block merge of 16 slots per class
  if (tid < 2) {
    const int c = tid;
    float Mx_ = -FLT_MAX, My_ = -FLT_MAX;
    for (int s2 = 0; s2 < 16; ++s2) {
      Mx_ = fmaxf(Mx_, st[c][s2][0]);
      My_ = fmaxf(My_, st[c][s2][4]);
    }
    float Z1x_ = 0, Z3x_ = 0, Wx_ = 0, Z1y_ = 0, Z3y_ = 0, Wy_ = 0;
    for (int s2 = 0; s2 < 16; ++s2) {
      float dx = st[c][s2][0] - Mx_;
      float sx1 = __expf(dx), sx3 = __expf(dx * (1.0f/3.0f));
      Z1x_ += st[c][s2][1] * sx1;
      Z3x_ += st[c][s2][2] * sx3;
      Wx_  += st[c][s2][3] * sx3;
      float dy = st[c][s2][4] - My_;
      float sy1 = __expf(dy), sy3 = __expf(dy * (1.0f/3.0f));
      Z1y_ += st[c][s2][5] * sy1;
      Z3y_ += st[c][s2][6] * sy3;
      Wy_  += st[c][s2][7] * sy3;
    }
    float* q = chunk_state + ((size_t)(b * SPLIT + chunk) * 16) + c * 8;
    q[0] = Mx_; q[1] = Z1x_; q[2] = Z3x_; q[3] = Wx_;
    q[4] = My_; q[5] = Z1y_; q[6] = Z3y_; q[7] = Wy_;
  }
}

// ------------------------------------------------------- finalize kernel ----
// 1 block x 128 threads; thread b merges its SPLIT chunk states, computes the
// four per-batch loss contributions, then thread 0 sums batches in fixed order.
__global__ void cm_finalize_kernel(const float* __restrict__ chunk_state,
                                   const float* __restrict__ heads,
                                   float* __restrict__ out)
{
  __shared__ float pb[4][B_SZ];
  const int b = threadIdx.x;

  float res[2][8];
  for (int c = 0; c < 2; ++c) {
    float Mx_ = -FLT_MAX, My_ = -FLT_MAX;
    for (int ch = 0; ch < SPLIT; ++ch) {
      const float* q = chunk_state + ((size_t)(b * SPLIT + ch) * 16) + c * 8;
      Mx_ = fmaxf(Mx_, q[0]); My_ = fmaxf(My_, q[4]);
    }
    float Z1x = 0, Z3x = 0, Wx = 0, Z1y = 0, Z3y = 0, Wy = 0;
    for (int ch = 0; ch < SPLIT; ++ch) {
      const float* q = chunk_state + ((size_t)(b * SPLIT + ch) * 16) + c * 8;
      float dx = q[0] - Mx_;
      float sx1 = __expf(dx), sx3 = __expf(dx * (1.0f/3.0f));
      Z1x += q[1] * sx1; Z3x += q[2] * sx3; Wx += q[3] * sx3;
      float dy = q[4] - My_;
      float sy1 = __expf(dy), sy3 = __expf(dy * (1.0f/3.0f));
      Z1y += q[5] * sy1; Z3y += q[6] * sy3; Wy += q[7] * sy3;
    }
    res[c][0] = Mx_; res[c][1] = Z1x; res[c][2] = Z3x; res[c][3] = Wx;
    res[c][4] = My_; res[c][5] = Z1y; res[c][6] = Z3y; res[c][7] = Wy;
  }

  // class0: x=a0(intra0), y=a1(intra1); class1: x=c01(cij), y=c10(cji)
  float lse1_a0  = res[0][0] + logf(res[0][1]);
  float lse3_a0  = res[0][0] * (1.0f/3.0f) + logf(res[0][2]);
  float lse1_a1  = res[0][4] + logf(res[0][5]);
  float lse3_a1  = res[0][4] * (1.0f/3.0f) + logf(res[0][6]);
  float lse1_c01 = res[1][0] + logf(res[1][1]);
  float lse3_c01 = res[1][0] * (1.0f/3.0f) + logf(res[1][2]);
  float lse1_c10 = res[1][4] + logf(res[1][5]);
  float lse3_c10 = res[1][4] * (1.0f/3.0f) + logf(res[1][6]);

  const float T = 3.0f;
  // KL(student s, teacher t) per batch = T^2*(W_t/(T*Z3_t) + lse3_s - lse3_t)
  float kd_a0_a1   = T * T * (res[0][7] / (T * res[0][6]) + lse3_a0  - lse3_a1);
  float kd_a1_a0   = T * T * (res[0][3] / (T * res[0][2]) + lse3_a1  - lse3_a0);
  float kd_c01_c10 = T * T * (res[1][7] / (T * res[1][6]) + lse3_c01 - lse3_c10);
  float kd_c10_c01 = T * T * (res[1][3] / (T * res[1][2]) + lse3_c10 - lse3_c01);

  pb[0][b] = (lse1_a0  - heads[b*4+0]) + (lse1_a1  - heads[b*4+1]);        // vcl
  pb[1][b] = kd_a0_a1 + kd_a1_a0;                                          // soft_vcl
  pb[2][b] = (lse1_c01 - 0.5f*heads[b*4+2]) + (lse1_c10 - 0.5f*heads[b*4+3]); // icl
  pb[3][b] = kd_c01_c10 + kd_c10_c01;                                      // soft_icl
  __syncthreads();

  if (b == 0) {
    float s0 = 0, s1 = 0, s2 = 0, s3 = 0;
    for (int i = 0; i < B_SZ; ++i) {
      s0 += pb[0][i]; s1 += pb[1][i]; s2 += pb[2][i]; s3 += pb[3][i];
    }
    const float invB = 1.0f / (float)B_SZ;
    out[0] = s0 * invB; out[1] = s1 * invB; out[2] = s2 * invB; out[3] = s3 * invB;
  }
}

// ------------------------------------------------------------ mem copy ------
__global__ void cm_copy_kernel(const float* __restrict__ mem0,
                               const float* __restrict__ mem1,
                               float* __restrict__ o0, float* __restrict__ o1)
{
  const size_t n4 = (size_t)N_MEM * D_DIM / 4;
  size_t i = (size_t)blockIdx.x * blockDim.x + threadIdx.x;
  const size_t stride = (size_t)gridDim.x * blockDim.x;
  const f4v* s0 = (const f4v*)mem0;
  const f4v* s1 = (const f4v*)mem1;
  f4v* d0 = (f4v*)o0;
  f4v* d1 = (f4v*)o1;
  for (; i < n4; i += stride) { d0[i] = s0[i]; d1[i] = s1[i]; }
}

// ----------------------------------------------- momentum row update --------
// single block, serialized over b -> deterministic last-write-wins on dup rows
__global__ void cm_update_kernel(const float* __restrict__ emb0,
                                 const float* __restrict__ emb1,
                                 const float* __restrict__ mem0,
                                 const float* __restrict__ mem1,
                                 const int* __restrict__ pos_idx,
                                 float* __restrict__ o0, float* __restrict__ o1)
{
  __shared__ float red[256];
  const int t = threadIdx.x;   // 128 threads, one per dim
  for (int b = 0; b < B_SZ; ++b) {
    int row = pos_idx[b * 2];
    float u0 = 0.5f * mem0[(size_t)row * D_DIM + t] + 0.5f * emb0[b * D_DIM + t];
    float u1 = 0.5f * mem1[(size_t)row * D_DIM + t] + 0.5f * emb1[b * D_DIM + t];
    red[t] = u0 * u0;
    red[128 + t] = u1 * u1;
    __syncthreads();
    for (int s = 64; s > 0; s >>= 1) {
      if (t < s) { red[t] += red[t + s]; red[128 + t] += red[128 + t + s]; }
      __syncthreads();
    }
    float n0 = sqrtf(red[0]);
    float n1 = sqrtf(red[128]);
    __syncthreads();
    o0[(size_t)row * D_DIM + t] = u0 / n0;
    o1[(size_t)row * D_DIM + t] = u1 / n1;
  }
}

// ---------------------------------------------------------------- launch ----
extern "C" void kernel_launch(void* const* d_in, const int* in_sizes, int n_in,
                              void* d_out, int out_size, void* d_ws, size_t ws_size,
                              hipStream_t stream)
{
  (void)in_sizes; (void)n_in; (void)out_size; (void)ws_size;
  const float* emb0 = (const float*)d_in[0];
  const float* emb1 = (const float*)d_in[1];
  const float* mem0 = (const float*)d_in[2];
  const float* mem1 = (const float*)d_in[3];
  const int*   pos  = (const int*)d_in[4];
  const int*   neg  = (const int*)d_in[5];
  float* out = (float*)d_out;

  float* chunk_state = (float*)d_ws;                       // B*SPLIT*16 floats
  float* heads       = chunk_state + B_SZ * SPLIT * 16;    // B*4 floats (~68KB total)

  cm_logits_kernel<<<B_SZ * SPLIT, 256, 0, stream>>>(emb0, emb1, mem0, mem1,
                                                     pos, neg, chunk_state, heads);
  cm_finalize_kernel<<<1, B_SZ, 0, stream>>>(chunk_state, heads, out);

  float* o0 = out + 4;
  float* o1 = o0 + (size_t)N_MEM * D_DIM;
  cm_copy_kernel<<<4096, 256, 0, stream>>>(mem0, mem1, o0, o1);
  cm_update_kernel<<<1, 128, 0, stream>>>(emb0, emb1, mem0, mem1, pos, o0, o1);
}
